// TrajectoryPredictorGRU_82051055223369
// MI455X (gfx1250) — matrix operand order repaired and checked by reading, not compiled, not run on hardware
//
#include <hip/hip_runtime.h>
#include <hip/hip_bf16.h>

typedef __attribute__((ext_vector_type(16))) _Float16 v16h;
typedef __attribute__((ext_vector_type(8)))  _Float16 v8h;
typedef __attribute__((ext_vector_type(8)))  float    v8f;
typedef __attribute__((ext_vector_type(4)))  float    v4f;

#define B_TOT   32768
#define T_SEQ   11
#define IN_D    2
#define H_D     64
#define N_STEPS 80
#define NT      12           // 192 / 16 output tiles
#define KTI     2            // 64 / 32 K tiles
#define RP      80           // padded A-staging row stride (halfs) -> 160B rows
#define WAVES   8
#define BLK_B   (WAVES * 16) // 128 batch rows per block

// ---- shared memory layout (bytes) ----
#define FRAG_HALFS 12288                       // per weight matrix: KTI*NT*32*16 halfs (24KB)
#define OFF_AST  (3 * FRAG_HALFS * 2)          // 73728, 16B aligned
#define AST_WAVE (2 * 16 * RP)                 // halfs per wave: c0 + c1 staging
#define OFF_PB   (OFF_AST + WAVES * AST_WAVE * 2)
#define SMEM_BYTES (OFF_PB + WAVES * 32 * 4)   // ~113.0 KB

// fast transcendentals: v_exp_f32 + v_rcp_f32 (both TRANS, co-execute with WMMA)
__device__ __forceinline__ float fastrcp(float x) { return __builtin_amdgcn_rcpf(x); }
__device__ __forceinline__ float sigmf(float x) { return fastrcp(1.0f + __expf(-x)); }
__device__ __forceinline__ float tanhfast(float x) {
  return fmaf(2.0f, fastrcp(1.0f + __expf(-2.0f * x)), -1.0f);
}

__device__ __forceinline__ v8f wmma16(v16h a, v16h b, v8f c) {
  return __builtin_amdgcn_wmma_f32_16x16x32_f16(false, a, false, b, (short)0, c, false, false);
}

// A fragment for 16x16x32: lane<16 -> row M=lane, halfs K = {32kt+0..7, 32kt+16..23}
//                          lane>=16 -> row M=lane-16, K = {32kt+8..15, 32kt+24..31}
__device__ __forceinline__ v16h loadA(const _Float16* st, int l15, int hf, int kt) {
  const _Float16* row = st + l15 * RP + kt * 32 + hf * 8;
  v8h lo = *(const v8h*)(row);
  v8h hi = *(const v8h*)(row + 16);
  return __builtin_shufflevector(lo, hi, 0, 1, 2, 3, 4, 5, 6, 7, 8, 9, 10, 11, 12, 13, 14, 15);
}

// B fragment: lane%16 = N column (output unit), 16 halfs = K values, same k order as A. Pre-packed in LDS.
__device__ __forceinline__ v16h loadB(const _Float16* wf, int lane, int kt, int nt) {
  const _Float16* f = wf + ((kt * NT + nt) * 32 + lane) * 16;
  v8h lo = *(const v8h*)(f);
  v8h hi = *(const v8h*)(f + 8);
  return __builtin_shufflevector(lo, hi, 0, 1, 2, 3, 4, 5, 6, 7, 8, 9, 10, 11, 12, 13, 14, 15);
}

// pack W [192][64] row-major fp32 -> B-fragment f16 layout:
// dst[((kt*12+nt)*32 + lane)*16 + j] = W[16*nt + lane%16][32*kt + 8*(lane/16) + (j&7) + 16*(j>>3)]
__device__ __forceinline__ void load_frag_matrix(_Float16* dst, const float* __restrict__ W, int tid) {
  for (int i = tid; i < FRAG_HALFS; i += 256) {
    int j  = i & 15;
    int r  = i >> 4;
    int L  = r & 31;
    int r2 = r >> 5;
    int nt = r2 % NT;
    int kt = r2 / NT;
    dst[i] = (_Float16)W[(16 * nt + (L & 15)) * H_D + (32 * kt + 8 * (L >> 4) + (j & 7) + 16 * (j >> 3))];
  }
}

// Per-lane, step-invariant scalar constants (registers, reloaded per phase).
struct GruConsts {
  // layer0 input weights (K=2) + biases; r/z biases pre-combined with b_hh
  float wr0[4], wr1[4], brt[4];
  float wz0[4], wz1[4], bzt[4];
  float wn0[4], wn1[4], bni[4];   // gi_n bias (b_ih only)
  float bhn0[4];                  // gh_n bias (b_hh only)
  // layer1 biases
  float brz1[8];                  // combined b_ih+b_hh for r (0..3) and z (4..7) tiles
  float bhn1[4], bin1[4];
};

__device__ __forceinline__ void load_consts(GruConsts& C,
                                            const float* __restrict__ Wih0,
                                            const float* __restrict__ bih0,
                                            const float* __restrict__ bhh0,
                                            const float* __restrict__ bih1,
                                            const float* __restrict__ bhh1,
                                            int l15) {
#pragma unroll
  for (int u = 0; u < 4; ++u) {
    int nr = 16 * u + l15, nz = 64 + nr, nn = 128 + nr;
    C.wr0[u] = Wih0[2 * nr]; C.wr1[u] = Wih0[2 * nr + 1]; C.brt[u] = bih0[nr] + bhh0[nr];
    C.wz0[u] = Wih0[2 * nz]; C.wz1[u] = Wih0[2 * nz + 1]; C.bzt[u] = bih0[nz] + bhh0[nz];
    C.wn0[u] = Wih0[2 * nn]; C.wn1[u] = Wih0[2 * nn + 1]; C.bni[u] = bih0[nn];
    C.bhn0[u] = bhh0[nn];
    C.bhn1[u] = bhh1[nn];
    C.bin1[u] = bih1[nn];
  }
#pragma unroll
  for (int g = 0; g < 8; ++g) {
    int n = 16 * g + l15;
    C.brz1[g] = bih1[n] + bhh1[n];
  }
}

// One fused 2-layer GRU step for this wave's 16 batch rows.
// D-layout registers: tile u, elem r -> (M = mbase+r, unit j = 16u + l15)
__device__ __forceinline__ void gru_step(const _Float16* __restrict__ wf0,   // W_hh0 frags
                                         const _Float16* __restrict__ wf1,   // W_ih1 frags
                                         const _Float16* __restrict__ wf2,   // W_hh1 frags
                                         int lane, int l15, int hf, int mbase,
                                         _Float16* c0st, _Float16* c1st, const float* pbuf,
                                         v8f h0t[4], v8f h1t[4], const GruConsts& C) {
  const v8f zro = {};
  v8f acc[12];

  // ---------------- layer 0: gh0 = h0 @ Whh0^T (WMMA, bias-free accums) ----------------
  {
    v16h a0 = loadA(c0st, l15, hf, 0);
    v16h a1 = loadA(c0st, l15, hf, 1);
#pragma unroll
    for (int nt = 0; nt < NT; ++nt) acc[nt] = wmma16(a0, loadB(wf0, lane, 0, nt), zro);
#pragma unroll
    for (int nt = 0; nt < NT; ++nt) acc[nt] = wmma16(a1, loadB(wf0, lane, 1, nt), acc[nt]);
  }

  float x0[8], x1[8];
#pragma unroll
  for (int r = 0; r < 8; ++r) { int m = mbase + r; x0[r] = pbuf[2 * m]; x1[r] = pbuf[2 * m + 1]; }

#pragma unroll
  for (int u = 0; u < 4; ++u) {
#pragma unroll
    for (int r = 0; r < 8; ++r) {
      float gir = fmaf(x1[r], C.wr1[u], fmaf(x0[r], C.wr0[u], C.brt[u]));
      float giz = fmaf(x1[r], C.wz1[u], fmaf(x0[r], C.wz0[u], C.bzt[u]));
      float gin = fmaf(x1[r], C.wn1[u], fmaf(x0[r], C.wn0[u], C.bni[u]));
      float rg = sigmf(gir + acc[u][r]);
      float zg = sigmf(giz + acc[4 + u][r]);
      float ng = tanhfast(fmaf(rg, acc[8 + u][r] + C.bhn0[u], gin));
      h0t[u][r] = fmaf(zg, h0t[u][r] - ng, ng);   // (1-z)*n + z*h, in place
    }
  }
  // stage new c0 as f16 A-matrix (used by layer1 gi below, layer0 gh next step)
#pragma unroll
  for (int u = 0; u < 4; ++u)
#pragma unroll
    for (int r = 0; r < 8; ++r)
      c0st[(mbase + r) * RP + 16 * u + l15] = (_Float16)h0t[u][r];

  // ---------------- layer 1 ----------------
  v8f accB[4];
  // gh1 first: depends only on old c1, hides the c0 store->load dependency
  {
    v16h a0 = loadA(c1st, l15, hf, 0);
    v16h a1 = loadA(c1st, l15, hf, 1);
#pragma unroll
    for (int nt = 0; nt < NT; ++nt) acc[nt] = wmma16(a0, loadB(wf2, lane, 0, nt), zro);
#pragma unroll
    for (int nt = 0; nt < NT; ++nt) acc[nt] = wmma16(a1, loadB(wf2, lane, 1, nt), acc[nt]);
  }
  // gi1: A = freshly staged c0; r/z chained onto acc, n kept separate in accB
  {
    v16h a0 = loadA(c0st, l15, hf, 0);
    v16h a1 = loadA(c0st, l15, hf, 1);
#pragma unroll
    for (int nt = 0; nt < 8; ++nt) acc[nt] = wmma16(a0, loadB(wf1, lane, 0, nt), acc[nt]);
#pragma unroll
    for (int u = 0; u < 4; ++u) accB[u] = wmma16(a0, loadB(wf1, lane, 0, 8 + u), zro);
#pragma unroll
    for (int nt = 0; nt < 8; ++nt) acc[nt] = wmma16(a1, loadB(wf1, lane, 1, nt), acc[nt]);
#pragma unroll
    for (int u = 0; u < 4; ++u) accB[u] = wmma16(a1, loadB(wf1, lane, 1, 8 + u), accB[u]);
  }
#pragma unroll
  for (int u = 0; u < 4; ++u) {
#pragma unroll
    for (int r = 0; r < 8; ++r) {
      float rg = sigmf(acc[u][r] + C.brz1[u]);
      float zg = sigmf(acc[4 + u][r] + C.brz1[4 + u]);
      float ng = tanhfast(fmaf(rg, acc[8 + u][r] + C.bhn1[u], accB[u][r] + C.bin1[u]));
      h1t[u][r] = fmaf(zg, h1t[u][r] - ng, ng);
    }
  }
#pragma unroll
  for (int u = 0; u < 4; ++u)
#pragma unroll
    for (int r = 0; r < 8; ++r)
      c1st[(mbase + r) * RP + 16 * u + l15] = (_Float16)h1t[u][r];
}

__global__ __launch_bounds__(256) void traj_gru_persistent_kernel(
    const float* __restrict__ x,
    const float* __restrict__ eWih0, const float* __restrict__ eWhh0,
    const float* __restrict__ ebih0, const float* __restrict__ ebhh0,
    const float* __restrict__ eWih1, const float* __restrict__ eWhh1,
    const float* __restrict__ ebih1, const float* __restrict__ ebhh1,
    const float* __restrict__ dWih0, const float* __restrict__ dWhh0,
    const float* __restrict__ dbih0, const float* __restrict__ dbhh0,
    const float* __restrict__ dWih1, const float* __restrict__ dWhh1,
    const float* __restrict__ dbih1, const float* __restrict__ dbhh1,
    const float* __restrict__ fcW,  const float* __restrict__ fcb,
    float* __restrict__ out) {
  extern __shared__ char smem[];
  const int tid = threadIdx.x;
  const int wave = tid >> 5, lane = tid & 31;
  const int l15 = lane & 15, hf = lane >> 4, mbase = hf * 8;
  const int batch0 = blockIdx.x * BLK_B + wave * 16;

  _Float16* wf0 = (_Float16*)smem;
  _Float16* wf1 = wf0 + FRAG_HALFS;
  _Float16* wf2 = wf0 + 2 * FRAG_HALFS;
  _Float16* c0st = (_Float16*)(smem + OFF_AST) + wave * AST_WAVE;
  _Float16* c1st = c0st + 16 * RP;
  float* pbuf = (float*)(smem + OFF_PB) + wave * 32;   // wave-private [m][p] layer0 input

  // zero the per-wave A staging (h0 = h1 = 0)
  {
    v4f z = {};
    v4f* p = (v4f*)c0st;
    for (int i = lane; i < (AST_WAVE * 2) / 16; i += 32) p[i] = z;
  }

  // ---- phase 0: encoder weights -> LDS fragments; scalars -> registers ----
  load_frag_matrix(wf0, eWhh0, tid);
  load_frag_matrix(wf1, eWih1, tid);
  load_frag_matrix(wf2, eWhh1, tid);
  GruConsts C;
  load_consts(C, eWih0, ebih0, ebhh0, ebih1, ebhh1, l15);
  __syncthreads();

  v8f h0t[4], h1t[4];
#pragma unroll
  for (int u = 0; u < 4; ++u)
#pragma unroll
    for (int r = 0; r < 8; ++r) { h0t[u][r] = 0.0f; h1t[u][r] = 0.0f; }

  // ---- encoder: T steps ----
  for (int t = 0; t < T_SEQ; ++t) {
    pbuf[l15 * 2 + hf] = x[(size_t)(batch0 + l15) * (T_SEQ * IN_D) + t * IN_D + hf];
    gru_step(wf0, wf1, wf2, lane, l15, hf, mbase, c0st, c1st, pbuf, h0t, h1t, C);
  }

  // ---- phase 1: decoder weights (overwrite LDS + registers) ----
  __syncthreads();
  load_frag_matrix(wf0, dWhh0, tid);
  load_frag_matrix(wf1, dWih1, tid);
  load_frag_matrix(wf2, dWhh1, tid);
  load_consts(C, dWih0, dbih0, dbhh0, dbih1, dbhh1, l15);
  __syncthreads();

  // fc head constants (registers)
  float fw0[4], fw1[4];
#pragma unroll
  for (int u = 0; u < 4; ++u) {
    fw0[u] = fcW[16 * u + l15];
    fw1[u] = fcW[64 + 16 * u + l15];
  }
  const float fb0 = fcb[0], fb1 = fcb[1];

  // decoder initial input = x[:, T-1, :]
  pbuf[l15 * 2 + hf] = x[(size_t)(batch0 + l15) * (T_SEQ * IN_D) + (T_SEQ - 1) * IN_D + hf];

  // ---- decoder: N_STEPS steps with FC head + feedback ----
  for (int s = 0; s < N_STEPS; ++s) {
    gru_step(wf0, wf1, wf2, lane, l15, hf, mbase, c0st, c1st, pbuf, h0t, h1t, C);
#pragma unroll
    for (int r = 0; r < 8; ++r) {
      float s0 = 0.0f, s1 = 0.0f;
#pragma unroll
      for (int u = 0; u < 4; ++u) {
        float c = h1t[u][r];
        s0 = fmaf(c, fw0[u], s0);
        s1 = fmaf(c, fw1[u], s1);
      }
#pragma unroll
      for (int off = 8; off >= 1; off >>= 1) {
        s0 += __shfl_xor(s0, off, 16);
        s1 += __shfl_xor(s1, off, 16);
      }
      if (l15 == 0) {
        float p0 = s0 + fb0, p1 = s1 + fb1;
        int m = mbase + r;
        size_t o = ((size_t)(batch0 + m) * N_STEPS + s) * 2;
        __builtin_nontemporal_store(p0, &out[o]);
        __builtin_nontemporal_store(p1, &out[o + 1]);
        pbuf[2 * m] = p0;       // feedback for next step's layer0 input
        pbuf[2 * m + 1] = p1;
      }
    }
  }
}

extern "C" void kernel_launch(void* const* d_in, const int* in_sizes, int n_in,
                              void* d_out, int out_size, void* d_ws, size_t ws_size,
                              hipStream_t stream) {
  (void)in_sizes; (void)n_in; (void)out_size; (void)d_ws; (void)ws_size;
  const float* x     = (const float*)d_in[0];
  const float* eWih0 = (const float*)d_in[1];
  const float* eWhh0 = (const float*)d_in[2];
  const float* ebih0 = (const float*)d_in[3];
  const float* ebhh0 = (const float*)d_in[4];
  const float* eWih1 = (const float*)d_in[5];
  const float* eWhh1 = (const float*)d_in[6];
  const float* ebih1 = (const float*)d_in[7];
  const float* ebhh1 = (const float*)d_in[8];
  const float* dWih0 = (const float*)d_in[9];
  const float* dWhh0 = (const float*)d_in[10];
  const float* dbih0 = (const float*)d_in[11];
  const float* dbhh0 = (const float*)d_in[12];
  const float* dWih1 = (const float*)d_in[13];
  const float* dWhh1 = (const float*)d_in[14];
  const float* dbih1 = (const float*)d_in[15];
  const float* dbhh1 = (const float*)d_in[16];
  const float* fcW   = (const float*)d_in[17];
  const float* fcb   = (const float*)d_in[18];
  float* out = (float*)d_out;

  (void)hipFuncSetAttribute((const void*)traj_gru_persistent_kernel,
                            hipFuncAttributeMaxDynamicSharedMemorySize, SMEM_BYTES);

  dim3 grid(B_TOT / BLK_B), block(256);
  traj_gru_persistent_kernel<<<grid, block, SMEM_BYTES, stream>>>(
      x, eWih0, eWhh0, ebih0, ebhh0, eWih1, eWhh1, ebih1, ebhh1,
      dWih0, dWhh0, dbih0, dbhh0, dWih1, dWhh1, dbih1, dbhh1,
      fcW, fcb, out);
}
